// GIN_44925357916335
// MI455X (gfx1250) — compile-verified, matrix operand order checked by reading, not run on hardware
//
#include <hip/hip_runtime.h>
#include <hip/hip_bf16.h>

// ---------------------------------------------------------------------------
// GIN (2x GINConv + global_add_pool) for MI455X / gfx1250, wave32 + WMMA.
//
// Shapes: N=100000 nodes, E=1200000 edges, G=256 graphs, all dims 64.
// Pipeline per layer: agg = scatter_add(x[src] -> dst);
//                     h   = relu((x+agg)@Wa + ba) @ Wb + bb   (fused WMMA MLP)
// Layer1 output gets an extra relu; layer2 output is written to d_out and
// atomically pooled per graph.
// ---------------------------------------------------------------------------

#define GIN_N 100000
#define GIN_E 1200000
#define GIN_G 256

typedef __bf16 v16bf __attribute__((ext_vector_type(16)));
typedef float  v8f   __attribute__((ext_vector_type(8)));

// -------------------------------- zero fill --------------------------------
__global__ void gin_zero_f32(float* __restrict__ p, long long n) {
    long long i = (long long)blockIdx.x * blockDim.x + threadIdx.x;
    long long stride = (long long)gridDim.x * blockDim.x;
    for (; i < n; i += stride) p[i] = 0.0f;
}

// ------------------------- weight pack (fp32 -> bf16 B-frag) ----------------
// For each weight matrix W[64][64] (row = k/fan_in, col = n/fan_out), emit the
// WMMA 16x16x32 bf16 B-fragment lane layout:
//   chunk id = ((w*4 + t)*2 + s)*32 + lane, 16 bf16 (32B) per chunk.
//   lane: n = t*16 + (lane&15); kb = (lane>>4)*8
//   element i<8 : k = s*32 + kb + i
//   element i>=8: k = s*32 + 16 + kb + (i-8)
__global__ void gin_pack_weights(const float* __restrict__ W1a,
                                 const float* __restrict__ W1b,
                                 const float* __restrict__ W2a,
                                 const float* __restrict__ W2b,
                                 __bf16* __restrict__ wpack) {
    const float* Ws[4] = {W1a, W1b, W2a, W2b};
    int idx = blockIdx.x * blockDim.x + threadIdx.x;   // 0..1023 chunk id
    if (idx >= 4 * 4 * 2 * 32) return;
    const int lane = idx & 31;
    const int s    = (idx >> 5) & 1;
    const int t    = (idx >> 6) & 3;
    const int w    = (idx >> 8) & 3;
    const float* W = Ws[w];
    const int n  = t * 16 + (lane & 15);
    const int kb = (lane >> 4) * 8;
    __bf16* out = wpack + (size_t)idx * 16;
#pragma unroll
    for (int i = 0; i < 16; ++i) {
        const int k = s * 32 + ((i < 8) ? (kb + i) : (16 + kb + (i - 8)));
        out[i] = (__bf16)W[k * 64 + n];
    }
}

// ------------------------------ edge scatter -------------------------------
// 16 threads per edge; each thread gathers a float4 of the source row and
// atomically accumulates it into the destination row (L2 atomics).
__global__ void gin_scatter_add(const float* __restrict__ feat,
                                const int* __restrict__ src,
                                const int* __restrict__ dst,
                                float* __restrict__ agg, int nedges) {
    long long gid = (long long)blockIdx.x * blockDim.x + threadIdx.x;
    if (gid >= (long long)nedges * 16) return;
    const int e = (int)(gid >> 4);
    const int q = (int)(gid & 15);
    const int s = src[e];
    const int d = dst[e];
    const float4 v = *(const float4*)(feat + (long long)s * 64 + q * 4);
    float* out = agg + (long long)d * 64 + q * 4;
    atomicAdd(out + 0, v.x);
    atomicAdd(out + 1, v.y);
    atomicAdd(out + 2, v.z);
    atomicAdd(out + 3, v.w);
}

// ------------------------- fused GIN MLP (WMMA) ----------------------------
// One wave handles 16 nodes. A fragments follow the ISA 16-bit 16x32 layout:
//   lane: m = lane&15, kb = (lane>>4)*8
//   elem i<8 : K = kb + i         elem i>=8: K = 16 + kb + (i-8)
// C/D layout: VGPR r, lane l -> row (r + 8*(l>>4)), col (l&15).
// The ReLU relayout between the two GEMMs goes through a per-wave LDS tile
// (stride 68 floats to spread banks).
__global__ void __launch_bounds__(256)
gin_mlp(const float* __restrict__ hin, const float* __restrict__ agg,
        const __bf16* __restrict__ wpackA, const __bf16* __restrict__ wpackB,
        const float* __restrict__ biasA, const float* __restrict__ biasB,
        float* __restrict__ hout, int relu_out,
        const int* __restrict__ batch, float* __restrict__ pooled,
        int ntiles) {
    __shared__ float lds[8 * 16 * 68];

    const int lane = threadIdx.x & 31;
    const int wave = threadIdx.x >> 5;
    const int m    = lane & 15;
    const int half = lane >> 4;          // 0 or 1
    const int kb   = half << 3;          // 0 or 8

    const int tile = blockIdx.x * 8 + wave;
    if (tile >= ntiles) return;          // wave-uniform; no block barriers used
    const long long rowBase = (long long)tile * 16;

    // ---- A fragments of h0 = hin + agg (fp32 -> bf16) ----
    const float* xr = hin + (rowBase + m) * 64;
    const float* gr = agg + (rowBase + m) * 64;
    float hv[32];
#pragma unroll
    for (int run = 0; run < 4; ++run) {
        const int c = run * 16 + kb;
        const float4 x0 = *(const float4*)(xr + c);
        const float4 x1 = *(const float4*)(xr + c + 4);
        const float4 g0 = *(const float4*)(gr + c);
        const float4 g1 = *(const float4*)(gr + c + 4);
        hv[run * 8 + 0] = x0.x + g0.x;  hv[run * 8 + 1] = x0.y + g0.y;
        hv[run * 8 + 2] = x0.z + g0.z;  hv[run * 8 + 3] = x0.w + g0.w;
        hv[run * 8 + 4] = x1.x + g1.x;  hv[run * 8 + 5] = x1.y + g1.y;
        hv[run * 8 + 6] = x1.z + g1.z;  hv[run * 8 + 7] = x1.w + g1.w;
    }
    v16bf a0, a1;
#pragma unroll
    for (int i = 0; i < 16; ++i) {
        a0[i] = (__bf16)hv[i];
        a1[i] = (__bf16)hv[16 + i];
    }

    // ---- GEMM1: (16x64) @ Wa, bias, relu -> LDS tile ----
    float* lbase = &lds[wave * 16 * 68];
#pragma unroll
    for (int t = 0; t < 4; ++t) {
        const v16bf b0 = *(const v16bf*)(wpackA + ((size_t)(t * 2 + 0) * 32 + lane) * 16);
        const v16bf b1 = *(const v16bf*)(wpackA + ((size_t)(t * 2 + 1) * 32 + lane) * 16);
        v8f c = {};
        c = __builtin_amdgcn_wmma_f32_16x16x32_bf16(false, a0, false, b0, (short)0, c, false, false);
        c = __builtin_amdgcn_wmma_f32_16x16x32_bf16(false, a1, false, b1, (short)0, c, false, false);
        const float bv = biasA[t * 16 + m];
#pragma unroll
        for (int r = 0; r < 8; ++r) {
            float f = c[r] + bv;
            f = f > 0.0f ? f : 0.0f;
            lbase[(r + half * 8) * 68 + t * 16 + m] = f;
        }
    }
    __builtin_amdgcn_wave_barrier();     // keep DS store->load ordering

    // ---- reload hidden tile in A-fragment layout ----
    const float* lr = lbase + m * 68;
    float hw[32];
#pragma unroll
    for (int run = 0; run < 4; ++run) {
        const int c = run * 16 + kb;
        const float4 v0 = *(const float4*)(lr + c);
        const float4 v1 = *(const float4*)(lr + c + 4);
        hw[run * 8 + 0] = v0.x;  hw[run * 8 + 1] = v0.y;
        hw[run * 8 + 2] = v0.z;  hw[run * 8 + 3] = v0.w;
        hw[run * 8 + 4] = v1.x;  hw[run * 8 + 5] = v1.y;
        hw[run * 8 + 6] = v1.z;  hw[run * 8 + 7] = v1.w;
    }
    v16bf h0, h1;
#pragma unroll
    for (int i = 0; i < 16; ++i) {
        h0[i] = (__bf16)hw[i];
        h1[i] = (__bf16)hw[16 + i];
    }

    // ---- GEMM2: bias, optional relu, store (+ optional graph pooling) ----
#pragma unroll
    for (int t = 0; t < 4; ++t) {
        const v16bf b0 = *(const v16bf*)(wpackB + ((size_t)(t * 2 + 0) * 32 + lane) * 16);
        const v16bf b1 = *(const v16bf*)(wpackB + ((size_t)(t * 2 + 1) * 32 + lane) * 16);
        v8f c = {};
        c = __builtin_amdgcn_wmma_f32_16x16x32_bf16(false, h0, false, b0, (short)0, c, false, false);
        c = __builtin_amdgcn_wmma_f32_16x16x32_bf16(false, h1, false, b1, (short)0, c, false, false);
        const float bv = biasB[t * 16 + m];
#pragma unroll
        for (int r = 0; r < 8; ++r) {
            float f = c[r] + bv;
            if (relu_out) f = f > 0.0f ? f : 0.0f;
            const long long row = rowBase + r + half * 8;
            hout[row * 64 + t * 16 + m] = f;
            if (pooled) {
                const int g = batch[row];
                atomicAdd(&pooled[(long long)g * 64 + t * 16 + m], f);
            }
        }
    }
}

// ------------------------------- launcher ----------------------------------
extern "C" void kernel_launch(void* const* d_in, const int* in_sizes, int n_in,
                              void* d_out, int out_size, void* d_ws, size_t ws_size,
                              hipStream_t stream) {
    (void)in_sizes; (void)n_in; (void)out_size; (void)ws_size;

    const float* x   = (const float*)d_in[0];
    const int*   ei  = (const int*)d_in[1];          // [2, E] flat: src then dst
    const int*   bat = (const int*)d_in[2];
    const float* W1a = (const float*)d_in[3];
    const float* b1a = (const float*)d_in[4];
    const float* W1b = (const float*)d_in[5];
    const float* b1b = (const float*)d_in[6];
    const float* W2a = (const float*)d_in[7];
    const float* b2a = (const float*)d_in[8];
    const float* W2b = (const float*)d_in[9];
    const float* b2b = (const float*)d_in[10];

    const int* src = ei;
    const int* dst = ei + GIN_E;

    float* pooled = (float*)d_out;                   // [G, 64]
    float* hout   = pooled + (size_t)GIN_G * 64;     // [N, 64]

    // workspace: agg (N*64 f32) | h1 (N*64 f32) | packed weights (16K bf16)
    float*  agg   = (float*)d_ws;
    float*  h1    = agg + (size_t)GIN_N * 64;
    __bf16* wpack = (__bf16*)(h1 + (size_t)GIN_N * 64);
    const size_t WP = (size_t)4 * 2 * 32 * 16;       // bf16 elems per matrix

    const int ntiles   = GIN_N / 16;                 // 6250
    const int mlpBlks  = (ntiles + 7) / 8;
    const long long scatWork = (long long)GIN_E * 16;
    const int scatBlks = (int)((scatWork + 255) / 256);
    const int zeroBlksN = (GIN_N * 64 + 255) / 256;

    // prep: zero agg + pooled, pack weights
    gin_zero_f32<<<zeroBlksN, 256, 0, stream>>>(agg, (long long)GIN_N * 64);
    gin_zero_f32<<<64, 256, 0, stream>>>(pooled, (long long)GIN_G * 64);
    gin_pack_weights<<<4, 256, 0, stream>>>(W1a, W1b, W2a, W2b, wpack);

    // layer 1
    gin_scatter_add<<<scatBlks, 256, 0, stream>>>(x, src, dst, agg, GIN_E);
    gin_mlp<<<mlpBlks, 256, 0, stream>>>(x, agg, wpack + 0 * WP, wpack + 1 * WP,
                                         b1a, b1b, h1, /*relu_out=*/1,
                                         nullptr, nullptr, ntiles);

    // layer 2 (+ pooling)
    gin_zero_f32<<<zeroBlksN, 256, 0, stream>>>(agg, (long long)GIN_N * 64);
    gin_scatter_add<<<scatBlks, 256, 0, stream>>>(h1, src, dst, agg, GIN_E);
    gin_mlp<<<mlpBlks, 256, 0, stream>>>(h1, agg, wpack + 2 * WP, wpack + 3 * WP,
                                         b2a, b2b, hout, /*relu_out=*/0,
                                         bat, pooled, ntiles);
}
